// LoadBalancedMoELayer_48524540510709
// MI455X (gfx1250) — matrix operand order repaired and checked by reading, not compiled, not run on hardware
//
#include <hip/hip_runtime.h>
#include <cstdint>
#include <cstddef>

#define NE      8
#define TOPK    2
#define AUXC    0.01f
#define ZC      0.001f
#define MT      64          // tokens per MoE tile
#define HID_MAX 2048

typedef __attribute__((ext_vector_type(8)))  __bf16 v8bf;
typedef __attribute__((ext_vector_type(16))) __bf16 v16bf;
typedef __attribute__((ext_vector_type(8)))  float  v8f;

__device__ __forceinline__ v16bf cat16(v8bf lo, v8bf hi) {
  return __builtin_shufflevector(lo, hi, 0,1,2,3,4,5,6,7,8,9,10,11,12,13,14,15);
}

// ---------------------------------------------------------------- conversions
__global__ void cvt_x_kernel(const float* __restrict__ src, __bf16* __restrict__ dst,
                             long long total) {
  long long i = (long long)blockIdx.x * blockDim.x + threadIdx.x;
  if (i < total) dst[i] = (__bf16)src[i];
}

// w: [E][K][Ncols] row-major  ->  wt: [E][Ncols][K]  (K becomes contiguous)
__global__ void cvt_transpose_kernel(const float* __restrict__ w, __bf16* __restrict__ wt,
                                     int K, int Ncols, long long total) {
  long long i = (long long)blockIdx.x * blockDim.x + threadIdx.x;
  if (i >= total) return;
  int n = (int)(i % Ncols);
  long long t = i / Ncols;
  int k = (int)(t % K);
  int e = (int)(t / K);
  wt[((long long)e * Ncols + n) * K + k] = (__bf16)w[i];
}

// ---------------------------------------------------------------- router
__global__ void router_kernel(const float* __restrict__ x, const float* __restrict__ rw,
                              int* __restrict__ cnt, int* __restrict__ toklist,
                              float* __restrict__ gatelist,
                              float* __restrict__ pbuf, float* __restrict__ zbuf,
                              int N, int D) {
  int tok  = (blockIdx.x * blockDim.x + threadIdx.x) >> 5;
  int lane = threadIdx.x & 31;
  if (tok >= N) return;
  const float* xr = x + (size_t)tok * D;

  float acc[NE];
#pragma unroll
  for (int e = 0; e < NE; ++e) acc[e] = 0.f;
  for (int d = lane; d < D; d += 32) {
    float xv = xr[d];
#pragma unroll
    for (int e = 0; e < NE; ++e) acc[e] = fmaf(xv, rw[e * D + d], acc[e]);
  }
#pragma unroll
  for (int off = 16; off > 0; off >>= 1) {
#pragma unroll
    for (int e = 0; e < NE; ++e) acc[e] += __shfl_xor(acc[e], off, 32);
  }
  if (lane == 0) {
    float mx = acc[0];
#pragma unroll
    for (int e = 1; e < NE; ++e) mx = fmaxf(mx, acc[e]);
    float p[NE], se = 0.f;
#pragma unroll
    for (int e = 0; e < NE; ++e) { p[e] = expf(acc[e] - mx); se += p[e]; }
    float inv = 1.f / se;
#pragma unroll
    for (int e = 0; e < NE; ++e) p[e] *= inv;
    float lse = mx + logf(se);
    zbuf[tok] = lse * lse;
#pragma unroll
    for (int e = 0; e < NE; ++e) pbuf[e * N + tok] = p[e];

    int i0 = 0;
#pragma unroll
    for (int e = 1; e < NE; ++e) if (p[e] > p[i0]) i0 = e;
    int i1 = (i0 == 0) ? 1 : 0;
#pragma unroll
    for (int e = 0; e < NE; ++e) if (e != i0 && p[e] > p[i1]) i1 = e;
    float s  = fmaxf(p[i0] + p[i1], 1e-9f);
    float g0 = p[i0] / s, g1 = p[i1] / s;

    int s0 = atomicAdd(&cnt[i0], 1);
    toklist[i0 * N + s0] = tok;  gatelist[i0 * N + s0] = g0;
    int s1 = atomicAdd(&cnt[i1], 1);
    toklist[i1 * N + s1] = tok;  gatelist[i1 * N + s1] = g1;
  }
}

// deterministic reductions: blocks 0..NE-1 sum pbuf rows, block NE sums zbuf
__global__ void reduce_kernel(const float* __restrict__ pbuf, const float* __restrict__ zbuf,
                              float* __restrict__ psum, float* __restrict__ zsum, int N) {
  __shared__ float sh[256];
  int b = blockIdx.x;
  const float* src = (b < NE) ? (pbuf + (size_t)b * N) : zbuf;
  float s = 0.f;
  for (int i = threadIdx.x; i < N; i += 256) s += src[i];
  sh[threadIdx.x] = s;
  __syncthreads();
  for (int st = 128; st > 0; st >>= 1) {
    if (threadIdx.x < st) sh[threadIdx.x] += sh[threadIdx.x + st];
    __syncthreads();
  }
  if (threadIdx.x == 0) {
    if (b < NE) psum[b] = sh[0]; else *zsum = sh[0];
  }
}

__global__ void finalize_kernel(const int* __restrict__ cnt, const float* __restrict__ psum,
                                const float* __restrict__ zsum, float* __restrict__ dst, int N) {
  if (blockIdx.x || threadIdx.x) return;
  float tot = (float)N * (float)TOPK;
  float aux = 0.f;
#pragma unroll
  for (int e = 0; e < NE; ++e)
    aux += ((float)cnt[e] / tot) * (psum[e] / (float)N);
  aux *= (float)NE;
  *dst = AUXC * aux + ZC * ((*zsum) / (float)N);
}

// ---------------------------------------------------------------- fused MoE MLP
__global__ __launch_bounds__(256, 1)
void moe_kernel(const __bf16* __restrict__ xb,
                const __bf16* __restrict__ w1t, const float* __restrict__ b1,
                const __bf16* __restrict__ w2t, const float* __restrict__ b2,
                const int* __restrict__ cnt, const int* __restrict__ toklist,
                const float* __restrict__ gatelist,
                float* __restrict__ out, int N, int D, int H) {
  __shared__ __bf16 h_lds[MT * HID_MAX];   // 256 KB: fused hidden tile, never hits HBM
  __shared__ int    tok_s[MT];
  __shared__ float  gate_s[MT];

  const int e     = blockIdx.y;
  const int row0  = blockIdx.x * MT;
  const int count = cnt[e];
  if (row0 >= count) return;
  const int rows = min(MT, count - row0);

  const int tid  = threadIdx.x;
  const int lane = tid & 31;
  const int wid  = tid >> 5;
  const int rg   = wid >> 2;          // 0..1 : 32 token rows each
  const int cg   = wid & 3;           // 0..3 : column groups

  if (tid < MT) {
    int r = row0 + tid;
    tok_s[tid]  = (tid < rows) ? toklist[(size_t)e * N + r] : 0;
    gate_s[tid] = (tid < rows) ? gatelist[(size_t)e * N + r] : 0.f;
  }
  __syncthreads();

  const int mrow  = lane & 15;               // A row / B col within a 16x16 tile
  const int ahalf = (lane < 16) ? 0 : 8;     // A-frag K sub-offset
  const int khalf = (lane < 16) ? 0 : 16;    // unused; B-frag also uses ahalf
  const int msel  = (lane < 16) ? 0 : 8;     // C-frag row offset

  const __bf16* xrow0 = xb + (size_t)tok_s[rg * 32 + mrow]      * D;
  const __bf16* xrow1 = xb + (size_t)tok_s[rg * 32 + 16 + mrow] * D;

  // ---------------- GEMM1: h = relu(x @ W1 + b1), kept in LDS as bf16 --------
  const int HB = H >> 2;                     // cols per col-group
  for (int sub = 0; sub < HB; sub += 64) {
    const int n0 = cg * HB + sub;
    v8f acc[2][4] = {};
    for (int k0 = 0; k0 < D; k0 += 32) {
      v16bf a0 = cat16(*(const v8bf*)(xrow0 + k0 + ahalf),
                       *(const v8bf*)(xrow0 + k0 + 16 + ahalf));
      v16bf a1 = cat16(*(const v8bf*)(xrow1 + k0 + ahalf),
                       *(const v8bf*)(xrow1 + k0 + 16 + ahalf));
#pragma unroll
      for (int j = 0; j < 4; ++j) {
        const __bf16* bp = w1t + ((size_t)e * H + (n0 + j * 16 + mrow)) * D + k0 + ahalf;
        __builtin_prefetch(bp + 64, 0, 0);
        v16bf bf = cat16(*(const v8bf*)bp, *(const v8bf*)(bp + 16));
        acc[0][j] = __builtin_amdgcn_wmma_f32_16x16x32_bf16(false, a0, false, bf,
                                                            (short)0, acc[0][j], false, false);
        acc[1][j] = __builtin_amdgcn_wmma_f32_16x16x32_bf16(false, a1, false, bf,
                                                            (short)0, acc[1][j], false, false);
      }
    }
#pragma unroll
    for (int i = 0; i < 2; ++i) {
#pragma unroll
      for (int j = 0; j < 4; ++j) {
        const int   coln = n0 + j * 16 + mrow;
        const float bias = b1[e * H + coln];
        const int   rbase = rg * 32 + i * 16 + msel;
#pragma unroll
        for (int v = 0; v < 8; ++v) {
          float hv = fmaxf(acc[i][j][v] + bias, 0.f);
          h_lds[(size_t)(rbase + v) * H + coln] = (__bf16)hv;
        }
      }
    }
  }
  __syncthreads();

  // ---------------- GEMM2: out += gate * (h @ W2 + b2) -----------------------
  const __bf16* hrow0 = h_lds + (size_t)(rg * 32 + mrow)      * H;
  const __bf16* hrow1 = h_lds + (size_t)(rg * 32 + 16 + mrow) * H;
  const int DB = D >> 2;
  for (int sub = 0; sub < DB; sub += 64) {
    const int n0 = cg * DB + sub;
    v8f acc[2][4] = {};
    for (int k0 = 0; k0 < H; k0 += 32) {
      v16bf a0 = cat16(*(const v8bf*)(hrow0 + k0 + ahalf),
                       *(const v8bf*)(hrow0 + k0 + 16 + ahalf));
      v16bf a1 = cat16(*(const v8bf*)(hrow1 + k0 + ahalf),
                       *(const v8bf*)(hrow1 + k0 + 16 + ahalf));
#pragma unroll
      for (int j = 0; j < 4; ++j) {
        const __bf16* bp = w2t + ((size_t)e * D + (n0 + j * 16 + mrow)) * H + k0 + ahalf;
        __builtin_prefetch(bp + 64, 0, 0);
        v16bf bf = cat16(*(const v8bf*)bp, *(const v8bf*)(bp + 16));
        acc[0][j] = __builtin_amdgcn_wmma_f32_16x16x32_bf16(false, a0, false, bf,
                                                            (short)0, acc[0][j], false, false);
        acc[1][j] = __builtin_amdgcn_wmma_f32_16x16x32_bf16(false, a1, false, bf,
                                                            (short)0, acc[1][j], false, false);
      }
    }
#pragma unroll
    for (int i = 0; i < 2; ++i) {
#pragma unroll
      for (int j = 0; j < 4; ++j) {
        const int   coln = n0 + j * 16 + mrow;
        const float bias = b2[e * D + coln];
        const int   rbase = rg * 32 + i * 16 + msel;
#pragma unroll
        for (int v = 0; v < 8; ++v) {
          const int r = rbase + v;
          if (r < rows) {
            float ov = gate_s[r] * (acc[i][j][v] + bias);
            atomicAdd(&out[(size_t)tok_s[r] * D + coln], ov);
          }
        }
      }
    }
  }
}

// ---------------------------------------------------------------- launch
extern "C" void kernel_launch(void* const* d_in, const int* in_sizes, int n_in,
                              void* d_out, int out_size, void* d_ws, size_t ws_size,
                              hipStream_t stream) {
  const float* x  = (const float*)d_in[0];
  const float* rw = (const float*)d_in[1];
  const float* w1 = (const float*)d_in[2];
  const float* b1 = (const float*)d_in[3];
  const float* w2 = (const float*)d_in[4];
  const float* b2 = (const float*)d_in[5];

  const int D = in_sizes[1] / NE;
  const int N = in_sizes[0] / D;
  const int H = in_sizes[3] / NE;
  float* out = (float*)d_out;

  // workspace carve-up
  char* ws = (char*)d_ws;
  int*   cnt      = (int*)ws;    ws += 256;
  float* psum     = (float*)ws;  ws += 256;
  float* zsum     = (float*)ws;  ws += 256;
  int*   toklist  = (int*)ws;    ws += sizeof(int)   * (size_t)NE * N;
  float* gatelist = (float*)ws;  ws += sizeof(float) * (size_t)NE * N;
  float* pbuf     = (float*)ws;  ws += sizeof(float) * (size_t)NE * N;
  float* zbuf     = (float*)ws;  ws += sizeof(float) * (size_t)N;
  ws = (char*)(((uintptr_t)ws + 255) & ~(uintptr_t)255);
  __bf16* xb  = (__bf16*)ws;     ws += sizeof(__bf16) * (size_t)N * D;
  __bf16* w1t = (__bf16*)ws;     ws += sizeof(__bf16) * (size_t)NE * D * H;
  __bf16* w2t = (__bf16*)ws;     ws += sizeof(__bf16) * (size_t)NE * H * D;
  if ((size_t)(ws - (char*)d_ws) > ws_size) return;  // workspace too small

  hipMemsetAsync(d_out, 0, sizeof(float) * (size_t)N * D, stream);
  hipMemsetAsync(cnt, 0, 256, stream);

  { long long tot = (long long)N * D;
    cvt_x_kernel<<<(unsigned)((tot + 255) / 256), 256, 0, stream>>>(x, xb, tot); }
  { long long tot = (long long)NE * D * H;
    cvt_transpose_kernel<<<(unsigned)((tot + 255) / 256), 256, 0, stream>>>(w1, w1t, D, H, tot);
    cvt_transpose_kernel<<<(unsigned)((tot + 255) / 256), 256, 0, stream>>>(w2, w2t, H, D, tot); }

  router_kernel<<<(N * 32 + 255) / 256, 256, 0, stream>>>(x, rw, cnt, toklist, gatelist,
                                                          pbuf, zbuf, N, D);
  reduce_kernel<<<NE + 1, 256, 0, stream>>>(pbuf, zbuf, psum, zsum, N);
  finalize_kernel<<<1, 32, 0, stream>>>(cnt, psum, zsum, out + (size_t)N * D, N);

  dim3 grid((N + MT - 1) / MT, NE);
  moe_kernel<<<grid, 256, 0, stream>>>(xb, w1t, b1, w2t, b2,
                                       cnt, toklist, gatelist, out, N, D, H);
}
